// ImplicitCGLayer_44298292691205
// MI455X (gfx1250) — matrix operand order repaired and checked by reading, not compiled, not run on hardware
//
#include <hip/hip_runtime.h>

#define B_ROWS 16384
#define FDIM   1024
#define TK     32

#define EPS_C   1e-8f
#define TOL_C   1e-6f
#define RIDGE_C 0.001f

typedef __bf16 v16bf __attribute__((ext_vector_type(16)));
typedef __bf16 v8bf  __attribute__((ext_vector_type(8)));
typedef float  v8f   __attribute__((ext_vector_type(8)));

union frag16 {
    v16bf v;
    v8bf  h[2];
};

__device__ __forceinline__ v8f wmma_bf16(v16bf a, v16bf b, v8f c) {
    return __builtin_amdgcn_wmma_f32_16x16x32_bf16(
        /*neg_a=*/false, a, /*neg_b=*/false, b,
        /*c_mod=*/(short)0, c, /*reuse_a=*/false, /*reuse_b=*/false);
}

// A-fragment in contiguous K order (matches B loads): lanes 0-15 hold K {0..15},
// lanes 16-31 hold K {16..31}. Both runs are contiguous 16B in a
// row-major [row][k] LDS tile -> two v8bf loads, zero VALU.
__device__ __forceinline__ v16bf load_afrag(const __bf16* row, int half) {
    frag16 u;
    u.h[0] = *(const v8bf*)(row + half * 16);
    u.h[1] = *(const v8bf*)(row + half * 16 + 8);
    return u.v;
}

// ---------------------------------------------------------------------------
// Elementwise f32 -> (bf16 hi, bf16 lo) split, grid-stride.
// ---------------------------------------------------------------------------
__global__ void __launch_bounds__(256)
split_f32(const float* __restrict__ in, __bf16* __restrict__ hi,
          __bf16* __restrict__ lo, int n) {
    for (int i = blockIdx.x * 256 + threadIdx.x; i < n; i += gridDim.x * 256) {
        const float f = in[i];
        const __bf16 h = (__bf16)f;
        hi[i] = h;
        lo[i] = (__bf16)(f - (float)h);
    }
}

// ---------------------------------------------------------------------------
// A = W^T W + ridge*I from pre-split Whi/Wlo; output stored as bf16 hi/lo.
// Grid: (F/64, F/64), block 128 (4 waves). Wave w: rows i0+w*16 x cols j0..+63.
// ---------------------------------------------------------------------------
__global__ void __launch_bounds__(128)
compute_A(const __bf16* __restrict__ Whi, const __bf16* __restrict__ Wlo,
          __bf16* __restrict__ Ahi, __bf16* __restrict__ Alo) {
    __shared__ __attribute__((aligned(32))) __bf16 ldsLh[64 * TK];  // [i][k]
    __shared__ __attribute__((aligned(32))) __bf16 ldsLl[64 * TK];
    __shared__ __attribute__((aligned(32))) __bf16 ldsRh[64 * TK];  // [j][k]
    __shared__ __attribute__((aligned(32))) __bf16 ldsRl[64 * TK];

    const int tid  = threadIdx.x;
    const int lane = tid & 31;
    const int wave = tid >> 5;
    const int half = lane >> 4;
    const int lc   = lane & 15;

    const int j0 = blockIdx.x * 64;
    const int i0 = blockIdx.y * 64;

    v8f acc[4];
    const v8f zero = {0.f,0.f,0.f,0.f,0.f,0.f,0.f,0.f};
#pragma unroll
    for (int t = 0; t < 4; ++t) acc[t] = zero;

    for (int k0 = 0; k0 < FDIM; k0 += TK) {
        // Stage transposed tiles: lds[c][k] = W[(k0+k)*F + base + c]
        for (int idx = tid; idx < TK * 64; idx += 128) {
            const int kk = idx >> 6;
            const int cc = idx & 63;
            const size_t g = (size_t)(k0 + kk) * FDIM;
            ldsLh[cc * TK + kk] = Whi[g + i0 + cc];
            ldsLl[cc * TK + kk] = Wlo[g + i0 + cc];
            ldsRh[cc * TK + kk] = Whi[g + j0 + cc];
            ldsRl[cc * TK + kk] = Wlo[g + j0 + cc];
        }
        __syncthreads();

        const int arow = (wave * 16 + lc) * TK;
        const v16bf ah = load_afrag(&ldsLh[arow], half);
        const v16bf al = load_afrag(&ldsLl[arow], half);

#pragma unroll
        for (int nt = 0; nt < 4; ++nt) {
            const int brow = (nt * 16 + lc) * TK + half * 16;
            const v16bf bh = *(const v16bf*)&ldsRh[brow];
            const v16bf bl = *(const v16bf*)&ldsRl[brow];
            acc[nt] = wmma_bf16(ah, bh, acc[nt]);
            acc[nt] = wmma_bf16(ah, bl, acc[nt]);
            acc[nt] = wmma_bf16(al, bh, acc[nt]);
        }
        __syncthreads();
    }

#pragma unroll
    for (int nt = 0; nt < 4; ++nt) {
        const int gj = j0 + nt * 16 + lc;
#pragma unroll
        for (int v = 0; v < 8; ++v) {
            const int gi = i0 + wave * 16 + half * 8 + v;
            float val = acc[nt][v];
            if (gi == gj) val += RIDGE_C;
            const __bf16 h = (__bf16)val;
            Ahi[(size_t)gi * FDIM + gj] = h;
            Alo[(size_t)gi * FDIM + gj] = (__bf16)(val - (float)h);
        }
    }
}

// ---------------------------------------------------------------------------
// Ap[b,n] = sum_k p[b,k] * A[n,k], operands pre-split in bf16 hi/lo.
// Grid: (F/128, B/128), block 256 (8 waves). Wave w = (wm,wn): 32M x 64N,
// 24 v_wmma per K-step, inner loop is ds_load + wmma only.
// ---------------------------------------------------------------------------
__global__ void __launch_bounds__(256)
cg_gemm_Ap(const __bf16* __restrict__ Ph, const __bf16* __restrict__ Pl,
           const __bf16* __restrict__ Ahi, const __bf16* __restrict__ Alo,
           float* __restrict__ C) {
    __shared__ __attribute__((aligned(32))) __bf16 ldsPh[128 * TK];  // [m][k]
    __shared__ __attribute__((aligned(32))) __bf16 ldsPl[128 * TK];
    __shared__ __attribute__((aligned(32))) __bf16 ldsAh[128 * TK];  // [n][k]
    __shared__ __attribute__((aligned(32))) __bf16 ldsAl[128 * TK];

    const int tid  = threadIdx.x;
    const int lane = tid & 31;
    const int wave = tid >> 5;
    const int half = lane >> 4;
    const int lc   = lane & 15;
    const int wn   = wave & 1;   // 2 N-groups of 64
    const int wm   = wave >> 1;  // 4 M-groups of 32

    const int n0 = blockIdx.x * 128;
    const int m0 = blockIdx.y * 128;

    v8f acc[2][4];
    const v8f zero = {0.f,0.f,0.f,0.f,0.f,0.f,0.f,0.f};
#pragma unroll
    for (int mt = 0; mt < 2; ++mt)
#pragma unroll
        for (int nt = 0; nt < 4; ++nt) acc[mt][nt] = zero;

    for (int k0 = 0; k0 < FDIM; k0 += TK) {
        // Stage 128x32 bf16 tiles (16B chunks, 2 per thread per tile)
        for (int idx = tid; idx < 128 * (TK / 8); idx += 256) {
            const int row = idx >> 2;
            const int c8  = (idx & 3) * 8;
            *(float4*)&ldsPh[row * TK + c8] =
                *(const float4*)&Ph[(size_t)(m0 + row) * FDIM + k0 + c8];
            *(float4*)&ldsPl[row * TK + c8] =
                *(const float4*)&Pl[(size_t)(m0 + row) * FDIM + k0 + c8];
            *(float4*)&ldsAh[row * TK + c8] =
                *(const float4*)&Ahi[(size_t)(n0 + row) * FDIM + k0 + c8];
            *(float4*)&ldsAl[row * TK + c8] =
                *(const float4*)&Alo[(size_t)(n0 + row) * FDIM + k0 + c8];
        }
        // Keep next K-slab of the L2-resident A warm
        if (k0 + TK < FDIM) {
            __builtin_prefetch(&Ahi[(size_t)(n0 + (tid >> 1)) * FDIM + k0 + TK], 0, 1);
            __builtin_prefetch(&Alo[(size_t)(n0 + (tid >> 1)) * FDIM + k0 + TK], 0, 1);
        }
        __syncthreads();

        v16bf ah[2], al[2];
#pragma unroll
        for (int mt = 0; mt < 2; ++mt) {
            const int arow = (wm * 32 + mt * 16 + lc) * TK;
            ah[mt] = load_afrag(&ldsPh[arow], half);
            al[mt] = load_afrag(&ldsPl[arow], half);
        }

#pragma unroll
        for (int nt = 0; nt < 4; ++nt) {
            const int brow = (wn * 64 + nt * 16 + lc) * TK + half * 16;
            const v16bf bh = *(const v16bf*)&ldsAh[brow];
            const v16bf bl = *(const v16bf*)&ldsAl[brow];
#pragma unroll
            for (int mt = 0; mt < 2; ++mt) {
                acc[mt][nt] = wmma_bf16(ah[mt], bh, acc[mt][nt]);  // hi*hi
                acc[mt][nt] = wmma_bf16(ah[mt], bl, acc[mt][nt]);  // hi*lo
                acc[mt][nt] = wmma_bf16(al[mt], bh, acc[mt][nt]);  // lo*hi
            }
        }
        __syncthreads();
    }

#pragma unroll
    for (int mt = 0; mt < 2; ++mt) {
        const int gm = m0 + wm * 32 + mt * 16 + half * 8;
#pragma unroll
        for (int nt = 0; nt < 4; ++nt) {
            const int gn = n0 + wn * 64 + nt * 16 + lc;
#pragma unroll
            for (int v = 0; v < 8; ++v)
                C[(size_t)(gm + v) * FDIM + gn] = acc[mt][nt][v];
        }
    }
}

// ---------------------------------------------------------------------------
// CG bookkeeping kernels (one block per batch row, F = 1024, 256 threads)
// ---------------------------------------------------------------------------
__global__ void __launch_bounds__(256)
cg_init(const float* __restrict__ xin, float* __restrict__ x,
        float* __restrict__ r, float* __restrict__ p,
        __bf16* __restrict__ ph, __bf16* __restrict__ pl,
        float* __restrict__ rs_old, int* __restrict__ done) {
    __shared__ float red[256];
    const int b = blockIdx.x;
    const size_t base = (size_t)b * FDIM;
    float s = 0.f;
    for (int j = threadIdx.x; j < FDIM; j += 256) {
        const float v = xin[base + j];
        x[base + j] = 0.f;
        r[base + j] = v;
        p[base + j] = v;
        const __bf16 h = (__bf16)v;
        ph[base + j] = h;
        pl[base + j] = (__bf16)(v - (float)h);
        s += v * v;
    }
    red[threadIdx.x] = s;
    __syncthreads();
    for (int off = 128; off > 0; off >>= 1) {
        if (threadIdx.x < off) red[threadIdx.x] += red[threadIdx.x + off];
        __syncthreads();
    }
    if (threadIdx.x == 0) {
        rs_old[b] = red[0];
        if (b == 0) *done = 0;
    }
}

__global__ void __launch_bounds__(256)
cg_alpha(const float* __restrict__ p, const float* __restrict__ Ap,
         const float* __restrict__ rs_old, float* __restrict__ alpha) {
    __shared__ float red[256];
    const int b = blockIdx.x;
    const size_t base = (size_t)b * FDIM;
    float s = 0.f;
    for (int j = threadIdx.x; j < FDIM; j += 256)
        s += p[base + j] * Ap[base + j];
    red[threadIdx.x] = s;
    __syncthreads();
    for (int off = 128; off > 0; off >>= 1) {
        if (threadIdx.x < off) red[threadIdx.x] += red[threadIdx.x + off];
        __syncthreads();
    }
    if (threadIdx.x == 0) alpha[b] = rs_old[b] / (red[0] + EPS_C);
}

__global__ void __launch_bounds__(256)
cg_update_xr(float* __restrict__ x, float* __restrict__ r,
             const float* __restrict__ p, const float* __restrict__ Ap,
             const float* __restrict__ alpha, float* __restrict__ rs_new,
             const int* __restrict__ done) {
    __shared__ float red[256];
    const int b = blockIdx.x;
    const bool d = (*done != 0);
    const float a = alpha[b];
    const size_t base = (size_t)b * FDIM;
    float s = 0.f;
    for (int j = threadIdx.x; j < FDIM; j += 256) {
        float rj = r[base + j];
        if (!d) {
            x[base + j] += a * p[base + j];
            rj -= a * Ap[base + j];
            r[base + j] = rj;
        }
        s += rj * rj;
    }
    red[threadIdx.x] = s;
    __syncthreads();
    for (int off = 128; off > 0; off >>= 1) {
        if (threadIdx.x < off) red[threadIdx.x] += red[threadIdx.x + off];
        __syncthreads();
    }
    if (threadIdx.x == 0) rs_new[b] = red[0];
}

__global__ void __launch_bounds__(1024)
cg_check(const float* __restrict__ rs_new, int* __restrict__ done) {
    __shared__ int red[1024];
    int ok = 1;
    for (int b = threadIdx.x; b < B_ROWS; b += 1024)
        ok &= (rs_new[b] < TOL_C) ? 1 : 0;
    red[threadIdx.x] = ok;
    __syncthreads();
    for (int off = 512; off > 0; off >>= 1) {
        if (threadIdx.x < off) red[threadIdx.x] &= red[threadIdx.x + off];
        __syncthreads();
    }
    if (threadIdx.x == 0 && red[0]) *done = 1;   // done |= all(rs_new < TOL)
}

__global__ void __launch_bounds__(256)
cg_update_p(float* __restrict__ p, __bf16* __restrict__ ph,
            __bf16* __restrict__ pl, const float* __restrict__ r,
            float* __restrict__ rs_old, const float* __restrict__ rs_new,
            const int* __restrict__ done) {
    const int b = blockIdx.x;
    if (*done != 0) return;   // uniform: p (and its split) stay unchanged
    const float beta = rs_new[b] / (rs_old[b] + EPS_C);
    const size_t base = (size_t)b * FDIM;
    for (int j = threadIdx.x; j < FDIM; j += 256) {
        const float v = r[base + j] + beta * p[base + j];
        p[base + j] = v;
        const __bf16 h = (__bf16)v;
        ph[base + j] = h;
        pl[base + j] = (__bf16)(v - (float)h);
    }
    __syncthreads();
    if (threadIdx.x == 0) rs_old[b] = rs_new[b];
}

// ---------------------------------------------------------------------------
extern "C" void kernel_launch(void* const* d_in, const int* in_sizes, int n_in,
                              void* d_out, int out_size, void* d_ws, size_t ws_size,
                              hipStream_t stream) {
    (void)in_sizes; (void)n_in; (void)out_size; (void)ws_size;
    const float* x_in   = (const float*)d_in[0];
    const float* weight = (const float*)d_in[1];
    float* x = (float*)d_out;

    char* ws = (char*)d_ws;
    const size_t BF = (size_t)B_ROWS * FDIM;
    const size_t FF = (size_t)FDIM * FDIM;
    size_t off = 0;
    float*  r      = (float*)(ws + off);  off += BF * sizeof(float);
    float*  p      = (float*)(ws + off);  off += BF * sizeof(float);
    float*  Ap     = (float*)(ws + off);  off += BF * sizeof(float);
    __bf16* ph     = (__bf16*)(ws + off); off += BF * 2;
    __bf16* pl     = (__bf16*)(ws + off); off += BF * 2;
    float*  rs_old = (float*)(ws + off);  off += B_ROWS * sizeof(float);
    float*  rs_new = (float*)(ws + off);  off += B_ROWS * sizeof(float);
    float*  alpha  = (float*)(ws + off);  off += B_ROWS * sizeof(float);
    int*    done   = (int*)(ws + off);    off += 64;
    __bf16* Ahi    = (__bf16*)(ws + off); off += FF * 2;
    __bf16* Alo    = (__bf16*)(ws + off); off += FF * 2;
    __bf16* Whi    = (__bf16*)(ws + off); off += FF * 2;
    __bf16* Wlo    = (__bf16*)(ws + off); off += FF * 2;

    // One-time: split W, build A = W^T W + ridge*I in split bf16 (L2-resident)
    split_f32<<<1024, 256, 0, stream>>>(weight, Whi, Wlo, (int)FF);
    dim3 gA(FDIM / 64, FDIM / 64);
    compute_A<<<gA, 128, 0, stream>>>(Whi, Wlo, Ahi, Alo);

    // x=0, r=p=x_in (+ split), rs_old = ||r||^2 per row, done=0
    cg_init<<<B_ROWS, 256, 0, stream>>>(x_in, x, r, p, ph, pl, rs_old, done);

    dim3 gG(FDIM / 128, B_ROWS / 128);
    for (int it = 0; it < 10; ++it) {
        cg_gemm_Ap<<<gG, 256, 0, stream>>>(ph, pl, Ahi, Alo, Ap);
        cg_alpha<<<B_ROWS, 256, 0, stream>>>(p, Ap, rs_old, alpha);
        cg_update_xr<<<B_ROWS, 256, 0, stream>>>(x, r, p, Ap, alpha, rs_new, done);
        cg_check<<<1, 1024, 0, stream>>>(rs_new, done);
        cg_update_p<<<B_ROWS, 256, 0, stream>>>(p, ph, pl, r, rs_old, rs_new, done);
    }
}
